// KMGenerator_89928025244535
// MI455X (gfx1250) — matrix operand order repaired and checked by reading, not compiled, not run on hardware
//
#include <hip/hip_runtime.h>

#define DIMK   768
#define KCENT  1024
#define NTOK   16384      // 16*1024 tokens per pair
#define MTILE  64         // tokens per block
#define NPASS  128        // centroids per pass
#define KC     32         // dim chunk (WMMA K)
#define LSTR   20         // LDS row stride in u32 (16 data + 4 pad): 80B, 16B-aligned,
                          // row*20 mod 64 distinct for 16 rows -> conflict-free b128 frag loads

typedef __attribute__((ext_vector_type(16))) __bf16 v16bf_t;
typedef __attribute__((ext_vector_type(8)))  float  v8f_t;

// fragment = 8 dwords; bf16 elems 2j/2j+1 live in dword j's lo/hi halves,
// matching the "2 consecutive k per u32" planes -> pure b128 copies, no shuffles.
union FragBF { v16bf_t v; unsigned u[8]; uint4 q[2]; };

// round-to-nearest-even bf16, kept in the HIGH 16 bits of a u32
__device__ __forceinline__ unsigned bf_rne_hi(unsigned u) {
    return (u + 0x7FFFu + ((u >> 16) & 1u)) & 0xFFFF0000u;
}
// pack two f32 -> one u32 of hi-bf16s and one u32 of residual-lo-bf16s (all 32-bit ops)
__device__ __forceinline__ void pack2_hilo(float x0, float x1, unsigned& wh, unsigned& wl) {
    unsigned r0 = bf_rne_hi(__float_as_uint(x0));
    unsigned r1 = bf_rne_hi(__float_as_uint(x1));
    wh = (r0 >> 16) | r1;
    float l0 = x0 - __uint_as_float(r0);
    float l1 = x1 - __uint_as_float(r1);
    unsigned s0 = bf_rne_hi(__float_as_uint(l0));
    unsigned s1 = bf_rne_hi(__float_as_uint(l1));
    wl = (s0 >> 16) | s1;
}

// ---------------- c_sq: ws[pair*1024 + n] = ||c_n||^2 -----------------------
__global__ __launch_bounds__(256) void csq_kernel(const float* __restrict__ c0,
                                                  const float* __restrict__ c1,
                                                  const float* __restrict__ c2,
                                                  float* __restrict__ ws) {
    const int wave = threadIdx.x >> 5, lane = threadIdx.x & 31;
    const int gi   = blockIdx.x * 8 + wave;          // 0..3071
    const int pair = gi >> 10, n = gi & 1023;
    const float* C = (pair == 0) ? c0 : (pair == 1) ? c1 : c2;
    const float* row = C + n * DIMK;
    float s = 0.f;
    for (int i = lane; i < DIMK; i += 32) { float x = row[i]; s += x * x; }
    #pragma unroll
    for (int off = 16; off; off >>= 1) s += __shfl_xor(s, off, 32);
    if (lane == 0) ws[pair * KCENT + n] = s;
}

// ---------------- fused GEMM (split-bf16 WMMA) + argmin ---------------------
__global__ __launch_bounds__(256)
void kmeans_kernel(const float* __restrict__ v0, const float* __restrict__ c0,
                   const float* __restrict__ v1, const float* __restrict__ c1,
                   const float* __restrict__ v2, const float* __restrict__ c2,
                   float* __restrict__ ws, float* __restrict__ out) {
    // double-buffered hi/lo planes: 2*(5120+5120+10240+10240) = 61440 B
    __shared__ unsigned lds_ah[2][MTILE * LSTR];
    __shared__ unsigned lds_al[2][MTILE * LSTR];
    __shared__ unsigned lds_bh[2][NPASS * LSTR];
    __shared__ unsigned lds_bl[2][NPASS * LSTR];
    __shared__ float red_val[MTILE * 2];
    __shared__ int   red_idx[MTILE * 2];
    __shared__ float vsq_tok[MTILE];
    __shared__ float tok_loss[MTILE];

    const int tid    = threadIdx.x;
    const int lane   = tid & 31, wave = tid >> 5;
    const int m_wave = wave & 3, n_wave = wave >> 2;       // 4 M-waves x 2 N-waves
    const int pair   = blockIdx.y;
    const int m_base = blockIdx.x * MTILE;

    const float* V   = (pair == 0) ? v0 : (pair == 1) ? v1 : v2;
    const float* C   = (pair == 0) ? c0 : (pair == 1) ? c1 : c2;
    const float* csq = ws + pair * KCENT;

    // cooperative-loader coordinates
    const int a_row = tid >> 2, a_col = (tid & 3) * 8;     // 64 rows x 8 f32 each
    const int b_row = tid >> 1, b_col = (tid & 1) * 16;    // 128 rows x 16 f32 each
    const float* Abase = V + (size_t)(m_base + a_row) * DIMK + a_col;

    // fragment coordinates (ISA 7.12.2 layouts)
    const int half = lane >> 4;
    const int lm   = lane & 15;
    const int a_c0 = half * 4;        // A frag: k = half*8 + i      -> quad at col half*4
    const int a_c1 = 8 + half * 4;    //         k = 16 + half*8 + i -> quad at col 8+half*4
    const int b_c0 = half * 4;        // B frag: same k order as A   -> quads at col half*4 and 8+half*4
    const int a_frag_row = (m_wave * 16 + lm) * LSTR;

    float min_s[8];
    int   min_i[8];
    #pragma unroll
    for (int r = 0; r < 8; ++r) { min_s[r] = 3.4e38f; min_i[r] = 0; }
    float vsq_part = 0.f;

    for (int pass = 0; pass < KCENT / NPASS; ++pass) {
        const float* Bbase = C + (size_t)(pass * NPASS + b_row) * DIMK + b_col;
        v8f_t acc[4] = {};
        int p = 0;

        // ---- staging closure: convert chunk [kb, kb+KC) into buffer `buf` ----
        auto stage = [&](int buf, int kb, bool accum_vsq) {
            {   // A: 64x32 f32 -> hi/lo planes (8 f32 per thread)
                const float* ag = Abase + kb;
                float4 x0 = ((const float4*)ag)[0];
                float4 x1 = ((const float4*)ag)[1];
                float xs[8] = {x0.x, x0.y, x0.z, x0.w, x1.x, x1.y, x1.z, x1.w};
                unsigned* dh = &lds_ah[buf][a_row * LSTR + (a_col >> 1)];
                unsigned* dl = &lds_al[buf][a_row * LSTR + (a_col >> 1)];
                #pragma unroll
                for (int j = 0; j < 4; ++j) {
                    float e0 = xs[2 * j], e1 = xs[2 * j + 1];
                    if (accum_vsq) vsq_part += e0 * e0 + e1 * e1;   // exact f32 ||v||^2
                    pack2_hilo(e0, e1, dh[j], dl[j]);
                }
                if (kb + KC < DIMK) __builtin_prefetch(ag + KC, 0, 1);  // global_prefetch_b8
            }
            {   // B: 128x32 f32 -> hi/lo planes (16 f32 per thread)
                const float* bg = Bbase + kb;
                float4 y0 = ((const float4*)bg)[0];
                float4 y1 = ((const float4*)bg)[1];
                float4 y2 = ((const float4*)bg)[2];
                float4 y3 = ((const float4*)bg)[3];
                float ys[16] = {y0.x, y0.y, y0.z, y0.w, y1.x, y1.y, y1.z, y1.w,
                                y2.x, y2.y, y2.z, y2.w, y3.x, y3.y, y3.z, y3.w};
                unsigned* dh = &lds_bh[buf][b_row * LSTR + (b_col >> 1)];
                unsigned* dl = &lds_bl[buf][b_row * LSTR + (b_col >> 1)];
                #pragma unroll
                for (int j = 0; j < 8; ++j)
                    pack2_hilo(ys[2 * j], ys[2 * j + 1], dh[j], dl[j]);
                if (kb + KC < DIMK) __builtin_prefetch(bg + KC, 0, 1);
            }
        };

        stage(p, 0, pass == 0);
        __syncthreads();

        for (int kb = 0; kb < DIMK; kb += KC) {
            // stage next chunk into the other buffer while this one is consumed
            if (kb + KC < DIMK) stage(p ^ 1, kb + KC, pass == 0);

            // A fragments: two 16B vector loads per plane, no unpacking
            FragBF ah, al;
            const unsigned* awh = &lds_ah[p][a_frag_row];
            const unsigned* awl = &lds_al[p][a_frag_row];
            ah.q[0] = *(const uint4*)(awh + a_c0);
            ah.q[1] = *(const uint4*)(awh + a_c1);
            al.q[0] = *(const uint4*)(awl + a_c0);
            al.q[1] = *(const uint4*)(awl + a_c1);

            #pragma unroll
            for (int t = 0; t < 4; ++t) {
                const int brow = (n_wave * 64 + t * 16 + lm) * LSTR + b_c0;
                FragBF bh, bl;
                bh.q[0] = *(const uint4*)(&lds_bh[p][brow]);
                bh.q[1] = *(const uint4*)(&lds_bh[p][brow + 8]);
                bl.q[0] = *(const uint4*)(&lds_bl[p][brow]);
                bl.q[1] = *(const uint4*)(&lds_bl[p][brow + 8]);
                // cross += ah*bh + ah*bl + al*bh  (split-fp32 precision)
                acc[t] = __builtin_amdgcn_wmma_f32_16x16x32_bf16(
                    false, ah.v, false, bh.v, (short)0, acc[t], false, false);
                acc[t] = __builtin_amdgcn_wmma_f32_16x16x32_bf16(
                    false, ah.v, false, bl.v, (short)0, acc[t], false, false);
                acc[t] = __builtin_amdgcn_wmma_f32_16x16x32_bf16(
                    false, al.v, false, bh.v, (short)0, acc[t], false, false);
            }
            __syncthreads();   // next chunk staged AND this buffer fully consumed
            p ^= 1;
        }

        if (pass == 0) {    // reduce the 4 loader lanes per token (lanes 4t..4t+3)
            float s = vsq_part;
            s += __shfl_xor(s, 1, 32);
            s += __shfl_xor(s, 2, 32);
            if ((tid & 3) == 0) vsq_tok[tid >> 2] = s;
        }

        // running min of s(n) = ||c_n||^2 - 2 v.c  (v^2 added later; argmin invariant)
        #pragma unroll
        for (int t = 0; t < 4; ++t) {
            const int n = pass * NPASS + n_wave * 64 + t * 16 + lm;
            const float cs = csq[n];
            #pragma unroll
            for (int r = 0; r < 8; ++r) {
                float s = __builtin_fmaf(-2.f, acc[t][r], cs);
                if (s < min_s[r]) { min_s[r] = s; min_i[r] = n; }  // n strictly increases: first-min kept
            }
        }
    }

    // cross-lane reduction within each 16-lane group (same token row)
    #pragma unroll
    for (int r = 0; r < 8; ++r) {
        float v = min_s[r];
        int   id = min_i[r];
        #pragma unroll
        for (int off = 1; off < 16; off <<= 1) {
            float ov = __shfl_xor(v, off, 32);
            int   oi = __shfl_xor(id, off, 32);
            if (ov < v || (ov == v && oi < id)) { v = ov; id = oi; }
        }
        if (lm == 0) {
            const int m_local = m_wave * 16 + half * 8 + r;   // C/D layout: M = r + 8*half
            red_val[m_local * 2 + n_wave] = v;
            red_idx[m_local * 2 + n_wave] = id;
        }
    }
    __syncthreads();

    if (tid < MTILE) {
        float va = red_val[tid * 2 + 0]; int ia = red_idx[tid * 2 + 0];
        float vb = red_val[tid * 2 + 1]; int ib = red_idx[tid * 2 + 1];
        float vm; int im;
        if (vb < va || (vb == va && ib < ia)) { vm = vb; im = ib; } else { vm = va; im = ia; }
        const int token = m_base + tid;
        out[1 + pair * NTOK + token] = (float)im;            // assignment
        tok_loss[tid] = vsq_tok[tid] + vm;                   // ||v||^2 + min_n s(n)
    }
    __syncthreads();
    if (tid == 0) {        // deterministic per-block partial (no float atomics)
        float s = 0.f;
        for (int i = 0; i < MTILE; ++i) s += tok_loss[i];
        ws[3 * KCENT + pair * 256 + blockIdx.x] = s;
    }
}

// ---------------- final scalar loss ----------------------------------------
__global__ void loss_kernel(const float* __restrict__ ws, float* __restrict__ out) {
    if (threadIdx.x == 0 && blockIdx.x == 0) {
        float tot = 0.f;
        for (int p = 0; p < 3; ++p) {
            float s = 0.f;
            for (int b = 0; b < 256; ++b) s += ws[3 * KCENT + p * 256 + b];
            tot += s / (float)NTOK;
        }
        out[0] = tot / 3.0f;
    }
}

extern "C" void kernel_launch(void* const* d_in, const int* in_sizes, int n_in,
                              void* d_out, int out_size, void* d_ws, size_t ws_size,
                              hipStream_t stream) {
    (void)in_sizes; (void)n_in; (void)out_size; (void)ws_size;
    const float* v0 = (const float*)d_in[0];
    const float* c0 = (const float*)d_in[1];
    const float* v1 = (const float*)d_in[2];
    const float* c1 = (const float*)d_in[3];
    const float* v2 = (const float*)d_in[4];
    const float* c2 = (const float*)d_in[5];
    float* ws  = (float*)d_ws;
    float* out = (float*)d_out;

    csq_kernel<<<384, 256, 0, stream>>>(c0, c1, c2, ws);
    dim3 grid(NTOK / MTILE, 3);
    kmeans_kernel<<<grid, 256, 0, stream>>>(v0, c0, v1, c1, v2, c2, ws, out);
    loss_kernel<<<1, 32, 0, stream>>>(ws, out);
}
